// MyModel_85298050498727
// MI455X (gfx1250) — matrix operand order repaired and checked by reading, not compiled, not run on hardware
//
#include <hip/hip_runtime.h>
#include <hip/hip_bf16.h>
#include <stdint.h>

// ---------------------------------------------------------------------------
// Problem geometry (fixed by the reference)
// ---------------------------------------------------------------------------
#define BATCH  128
#define CH     2
#define NROW   256
#define DDIM   256
#define KDIM   (CH*NROW*DDIM)     // 131072 : GEMM K
#define NCOL   256                // GEMM N (output cols of W)
#define KSEL   128                // rows kept per (b,c)
#define DROP_SCALE 1e-4f

// GEMM tiling
#define KCHUNK 2048               // K per block (split-K)
#define NKCH   (KDIM/KCHUNK)      // 64 K-chunks
#define NTILE  32                 // output cols per block
#define NJT    (NCOL/NTILE)       // 8 col tiles
#define KSTEP  32                 // bf16 WMMA K
#define NSTEP  (KCHUNK/KSTEP)     // 64 inner steps
#define AROW   36                 // padded LDS row stride in bf16 units (72 B)

typedef __attribute__((ext_vector_type(16))) __bf16   v16bf_t;
typedef __attribute__((ext_vector_type(8)))  float    v8f_t;
typedef __attribute__((ext_vector_type(8)))  uint32_t v8u_t;
typedef __attribute__((ext_vector_type(4)))  uint32_t v4u_t;
typedef __attribute__((ext_vector_type(8)))  int      v8i_t;
typedef __attribute__((ext_vector_type(4)))  int      v4i_t;

__device__ __forceinline__ uint16_t f2bf(float f) {
  // round-to-nearest-even f32 -> bf16
  uint32_t u = __builtin_bit_cast(uint32_t, f);
  uint32_t r = u + 0x7FFFu + ((u >> 16) & 1u);
  return (uint16_t)(r >> 16);
}

// ---------------------------------------------------------------------------
// TDM: async-load one 32(k) x 32(n) f32 tile of W (row stride NCOL) into LDS.
// D# per CDNA5 ISA 8.3/8.4: group0 = {flags, lds_addr, global_addr, type=2},
// group1 = {mask=0, data_size=4B, dims/strides}. Groups 2/3 zero (2-D tile).
// Issued by one wave; completion tracked with TENSORcnt (in-order).
// This toolchain declares the 6-arg builtin:
//   (uint32x4 g0, int32x8 g1, int32x4 g2, int32x4 g3, int32x8 extra, i32 cpol)
// ---------------------------------------------------------------------------
__device__ __forceinline__ void tdm_load_w_tile(const float* gptr, uint32_t lds_off) {
  const uint64_t ga = (uint64_t)(uintptr_t)gptr;
  v4u_t g0;
  g0[0] = 1u;                                    // count=1, user load descriptor
  g0[1] = lds_off;                               // LDS byte offset of dest tile
  g0[2] = (uint32_t)ga;                          // global_addr[31:0]
  g0[3] = (uint32_t)((ga >> 32) & 0x01FFFFFFull) // global_addr[56:32]
        | (2u << 30);                            // type = 2 ("image")
  v8i_t g1;
  g1[0] = (int)(2u << 16);                       // workgroup_mask=0, data_size=4B
  g1[1] = (int)((uint32_t)(NCOL & 0xFFFF) << 16);// tensor_dim0[15:0]=256 @bit48
  g1[2] = (int)((NCOL >> 16) | ((KDIM & 0xFFFF) << 16));     // dim0 hi | dim1 lo
  g1[3] = (int)((KDIM >> 16) | (32u << 16));     // tensor_dim1 hi | tile_dim0=32
  g1[4] = 32;                                    // tile_dim1=32, tile_dim2=0
  g1[5] = NCOL;                                  // tensor_dim0_stride[31:0]=256
  g1[6] = 0;                                     // stride hi | dim1_stride lo
  g1[7] = 0;
  const v4i_t z4 = {0, 0, 0, 0};
  const v8i_t z8 = {0, 0, 0, 0, 0, 0, 0, 0};
  __builtin_amdgcn_tensor_load_to_lds(g0, g1, z4, z4, z8, 0);
}

// ---------------------------------------------------------------------------
// Kernel 0: zero line_pre accumulator (32768 f32) + idx_att/idx_drp (32768 i32)
// ---------------------------------------------------------------------------
__global__ __launch_bounds__(256) void k_zero_ws(uint32_t* __restrict__ ws) {
  ws[blockIdx.x * 256 + threadIdx.x] = 0u;   // grid = 256 blocks
}

// ---------------------------------------------------------------------------
// Kernel 1: split-K GEMM  line_pre[b, j] += xf[b, :]chunk @ W[:, j]chunk
//   xf[b, k] = x[b, c, n, d]  with k = n*(D*C) + d*C + c  (channels-last flatten)
//   A: VGPR-staged + f32->bf16 convert (channel interleave) into LDS.
//   B: TDM tensor_load_to_lds double-buffered f32 tiles (async W stream),
//      converted to bf16 at fragment build.
// ---------------------------------------------------------------------------
__global__ __launch_bounds__(256) void k_gemm_wmma(const float* __restrict__ x,
                                                   const float* __restrict__ W,
                                                   float* __restrict__ line_pre) {
  __shared__ uint16_t As[BATCH * AROW];   // 128 x 32 bf16 tile, k-major rows
  __shared__ float    Bw[2][KSTEP * NTILE]; // double-buffered 32x32 f32 W tiles

  const int t    = threadIdx.x;
  const int wid  = t >> 5;               // wave32: 8 waves per block
  const int lane = t & 31;
  const int jt   = blockIdx.x & (NJT - 1);
  const int kc   = blockIdx.x / NJT;
  const int nj0  = jt * NTILE;

  v8f_t acc0 = {};                       // 16x16 tile at cols [nj0, nj0+16)
  v8f_t acc1 = {};                       // 16x16 tile at cols [nj0+16, nj0+32)

  // prologue: wave 0 launches the first W tile
  if (t < 32) {
    tdm_load_w_tile(W + (size_t)(kc * KCHUNK) * NCOL + nj0,
                    (uint32_t)(uintptr_t)&Bw[0][0]);
  }

  for (int s = 0; s < NSTEP; ++s) {
    const int kg0 = kc * KCHUNK + s * KSTEP;       // global k of this step
    const int n_g = kg0 >> 9;                      // n index (fixed per step)
    const int d0  = (kg0 & 511) >> 1;              // d range start (16 wide)

    // ---- wave 0: keep the TDM pipeline one tile ahead, then drain tile s --
    if (t < 32) {
      if (s + 1 < NSTEP) {
        tdm_load_w_tile(W + (size_t)(kg0 + KSTEP) * NCOL + nj0,
                        (uint32_t)(uintptr_t)&Bw[(s + 1) & 1][0]);
        __builtin_amdgcn_s_wait_tensorcnt(1);   // tile s landed (in-order)
      } else {
        __builtin_amdgcn_s_wait_tensorcnt(0);
      }
    }

    // ---- stage A: 128 batches x 32 k (d-range of 16, both channels) -------
    #pragma unroll
    for (int i = 0; i < 2; ++i) {
      const int lin = i * 256 + t;
      const int b   = lin >> 2;
      const int q   = lin & 3;
      const int d   = d0 + q * 4;
      const float* p0 = x + (((size_t)(b*2 + 0)*NROW + n_g)*DDIM + d);
      const float* p1 = x + (((size_t)(b*2 + 1)*NROW + n_g)*DDIM + d);
      const float4 a0 = *(const float4*)p0;
      const float4 a1 = *(const float4*)p1;
      __builtin_prefetch(p0 + 16, 0, 1);           // next step's A quad
      __builtin_prefetch(p1 + 16, 0, 1);
      uint16_t* dst = As + b * AROW + q * 8;       // k = 8q .. 8q+7
      uint2 pk0, pk1;
      pk0.x = (uint32_t)f2bf(a0.x) | ((uint32_t)f2bf(a1.x) << 16);
      pk0.y = (uint32_t)f2bf(a0.y) | ((uint32_t)f2bf(a1.y) << 16);
      pk1.x = (uint32_t)f2bf(a0.z) | ((uint32_t)f2bf(a1.z) << 16);
      pk1.y = (uint32_t)f2bf(a0.w) | ((uint32_t)f2bf(a1.w) << 16);
      *(uint2*)(dst)     = pk0;
      *(uint2*)(dst + 4) = pk1;
    }
    __syncthreads();   // A stores visible; TDM tile s drained by wave 0

    // ---- build fragments per ISA 7.12.2 lane layouts ----------------------
    const int mh   = lane & 15;
    const int half = lane >> 4;
    const uint16_t* arow = As + (wid * 16 + mh) * AROW;
    const float*    bw   = &Bw[s & 1][0];          // [k][n] row-major f32
    v8u_t au, bu0, bu1;
    #pragma unroll
    for (int r = 0; r < 4; ++r) {
      au[r]     = *(const uint32_t*)(arow + 8*half + 2*r);        // K 0..15 half
      au[4 + r] = *(const uint32_t*)(arow + 16 + 8*half + 2*r);   // K 16..31 half
    }
    #pragma unroll
    for (int r = 0; r < 8; ++r) {
      const int k = 8*half + 2*r + (r >= 4 ? 8 : 0);
      const float lo0 = bw[k * NTILE + mh];
      const float hi0 = bw[(k + 1) * NTILE + mh];
      const float lo1 = bw[k * NTILE + 16 + mh];
      const float hi1 = bw[(k + 1) * NTILE + 16 + mh];
      bu0[r] = (uint32_t)f2bf(lo0) | ((uint32_t)f2bf(hi0) << 16);
      bu1[r] = (uint32_t)f2bf(lo1) | ((uint32_t)f2bf(hi1) << 16);
    }
    const v16bf_t af = __builtin_bit_cast(v16bf_t, au);
    acc0 = __builtin_amdgcn_wmma_f32_16x16x32_bf16(
        false, af, false, __builtin_bit_cast(v16bf_t, bu0),
        (short)0, acc0, false, false);
    acc1 = __builtin_amdgcn_wmma_f32_16x16x32_bf16(
        false, af, false, __builtin_bit_cast(v16bf_t, bu1),
        (short)0, acc1, false, false);
    __syncthreads();   // done reading As / Bw[s&1] before they are rewritten
  }

  // ---- split-K reduction via f32 global atomics ---------------------------
  const int n    = lane & 15;
  const int half = lane >> 4;
  #pragma unroll
  for (int r = 0; r < 8; ++r) {
    const int m = wid * 16 + half * 8 + r;         // C/D layout: M = r + 8*half
    atomicAdd(line_pre + (size_t)m * NCOL + nj0 + n,      acc0[r]);
    atomicAdd(line_pre + (size_t)m * NCOL + nj0 + 16 + n, acc1[r]);
  }
}

// ---------------------------------------------------------------------------
// Kernel 2: per-batch median mask + stable index compaction.
// sigmoid is strictly monotone, so the mask computed on pre-activations with
// the midpoint of the two middle order statistics is identical to the
// reference's sigmoid-space mask.
// ---------------------------------------------------------------------------
__global__ __launch_bounds__(256) void k_median_idx(const float* __restrict__ line_pre,
                                                    int* __restrict__ idx_att,
                                                    int* __restrict__ idx_drp) {
  __shared__ float z[NROW];
  __shared__ int   msk[NROW];
  __shared__ float mid[2];
  const int b = blockIdx.x;
  const int t = threadIdx.x;

  const float zt = line_pre[(size_t)b * NROW + t];
  z[t] = zt;
  __syncthreads();

  // exact rank by counting (tie-broken by index -> a permutation of 0..255)
  int rank = 0;
  for (int j = 0; j < NROW; ++j) {
    const float zj = z[j];
    rank += (zj < zt) || (zj == zt && j < t);
  }
  if (rank == NROW/2 - 1) mid[0] = zt;   // lower middle
  if (rank == NROW/2)     mid[1] = zt;   // upper middle
  __syncthreads();

  const float med = 0.5f * (mid[0] + mid[1]);
  const int keep = (zt > med) ? 1 : 0;   // reference: med < line (strict)
  msk[t] = keep;
  __syncthreads();

  int pos = 0;                           // stable prefix over kept flags
  for (int j = 0; j < t; ++j) pos += msk[j];
  if (keep) {
    if (pos < KSEL) idx_att[b * KSEL + pos] = t;
  } else {
    const int p2 = t - pos;
    if (p2 < KSEL) idx_drp[b * KSEL + p2] = t;
  }
}

// ---------------------------------------------------------------------------
// Kernel 3: gather + epilogue  out[b,c,i,:] = x[b,c,att,:] + 1e-4*x[b,c,drp,:]
// 4 output rows per block, float4 vectorized (fully coalesced).
// ---------------------------------------------------------------------------
__global__ __launch_bounds__(256) void k_gather(const float* __restrict__ x,
                                                const int* __restrict__ idx_att,
                                                const int* __restrict__ idx_drp,
                                                float* __restrict__ out) {
  const int t    = threadIdx.x;
  const int r    = t >> 6;                         // row within block (0..3)
  const int lane = t & 63;                         // 64 float4 per 256-f32 row
  const int row  = blockIdx.x * 4 + r;             // 0 .. B*C*KSEL-1
  const int b    = row >> 8;                       // CH*KSEL = 256 rows / batch
  const int rem  = row & 255;
  const int c    = rem >> 7;
  const int i    = rem & 127;
  const int ia   = idx_att[b * KSEL + i] & 255;    // clamp: safety vs stale ws
  const int id   = idx_drp[b * KSEL + i] & 255;

  const float4* __restrict__ x4 = (const float4*)x;
  float4* __restrict__ o4 = (float4*)out;
  const size_t plane = ((size_t)b * CH + c) * NROW;
  const float4 va = x4[(plane + ia) * (DDIM/4) + lane];
  const float4 vd = x4[(plane + id) * (DDIM/4) + lane];
  float4 o;
  o.x = va.x + DROP_SCALE * vd.x;
  o.y = va.y + DROP_SCALE * vd.y;
  o.z = va.z + DROP_SCALE * vd.z;
  o.w = va.w + DROP_SCALE * vd.w;
  o4[(((size_t)b * CH + c) * KSEL + i) * (DDIM/4) + lane] = o;
}

// ---------------------------------------------------------------------------
// Host-side launcher
// ---------------------------------------------------------------------------
extern "C" void kernel_launch(void* const* d_in, const int* in_sizes, int n_in,
                              void* d_out, int out_size, void* d_ws, size_t ws_size,
                              hipStream_t stream) {
  (void)in_sizes; (void)n_in; (void)out_size; (void)ws_size;
  const float* x = (const float*)d_in[0];   // [128, 2, 256, 256] f32
  const float* W = (const float*)d_in[1];   // [131072, 256] f32
  float* out = (float*)d_out;               // [128, 2, 128, 256] f32

  // workspace layout (contiguous dwords; re-zeroed every call):
  float* line_pre = (float*)d_ws;                                    // 128*256 f32
  int*   idx_att  = (int*)((char*)d_ws + BATCH * NCOL * 4);          // 128*128 i32
  int*   idx_drp  = (int*)((char*)idx_att + BATCH * KSEL * 4);       // 128*128 i32

  k_zero_ws  <<<256, 256, 0, stream>>>((uint32_t*)d_ws);             // 65536 dwords
  k_gemm_wmma<<<NJT * NKCH, 256, 0, stream>>>(x, W, line_pre);       // 512 blocks
  k_median_idx<<<BATCH, 256, 0, stream>>>(line_pre, idx_att, idx_drp);
  k_gather   <<<BATCH * CH * KSEL / 4, 256, 0, stream>>>(x, idx_att, idx_drp, out);
}
